// GDEM_54047868453243
// MI455X (gfx1250) — matrix operand order repaired and checked by reading, not compiled, not run on hardware
//
#include <hip/hip_runtime.h>
#include <hip/hip_bf16.h>
#include <cstdint>

// Problem constants (from reference setup_inputs)
#define B_   2
#define C_   2048
#define H_   16
#define Wd_  16
#define HW_  256      // H*W
#define HP_  24       // padded spatial
#define P_   576      // 24*24 padded pixels
#define NH_  8
#define HD_  256      // C/NH
#define HID_ 512      // MLP hidden

typedef __attribute__((ext_vector_type(16))) _Float16 v16h;
typedef __attribute__((ext_vector_type(8)))  _Float16 f16x8;
typedef __attribute__((ext_vector_type(8)))  float    v8f;

static __device__ __forceinline__ v16h mk16(f16x8 lo, f16x8 hi) {
  return __builtin_shufflevector(lo, hi, 0,1,2,3,4,5,6,7,8,9,10,11,12,13,14,15);
}
static __device__ __forceinline__ v8f wmma16(v16h a, v16h b, v8f c) {
  return __builtin_amdgcn_wmma_f32_16x16x32_f16(false, a, false, b, (short)0, c, false, false);
}
static __device__ __forceinline__ int imin(int a, int b) { return a < b ? a : b; }

// ---------------- f32 -> f16 weight conversion ----------------
__global__ void cvt_f16_kernel(const float* __restrict__ in, _Float16* __restrict__ out, int n) {
  int i = blockIdx.x * blockDim.x + threadIdx.x;
  if (i < n) out[i] = (_Float16)in[i];
}

// ---------------- BN1 + reflect pad, pixel-major f16 ----------------
// ypad: (B, 576, C)  yint: (B, 256, C)
__global__ void bn_pad_kernel(const float* __restrict__ x,
                              const float* __restrict__ g, const float* __restrict__ bb,
                              const float* __restrict__ mm, const float* __restrict__ vv,
                              _Float16* __restrict__ ypad, _Float16* __restrict__ yint) {
  int p = blockIdx.x;            // padded pixel 0..575
  int b = blockIdx.y;
  int r = p / HP_, s = p % HP_;
  int i = r - 4, j = s - 4;
  int ri = (i < 0) ? -i : ((i > 15) ? 30 - i : i);   // torch 'reflect'
  int sj = (j < 0) ? -j : ((j > 15) ? 30 - j : j);
  bool interior = (i >= 0 && i < H_ && j >= 0 && j < Wd_);
  #pragma unroll
  for (int t = 0; t < C_ / 256; ++t) {
    int c = threadIdx.x + t * 256;
    float sc = g[c] * rsqrtf(vv[c] + 1e-5f);
    float sh = bb[c] - mm[c] * sc;
    float val = x[(((size_t)b * C_ + c) * H_ + ri) * Wd_ + sj] * sc + sh;
    ypad[((size_t)b * P_ + p) * C_ + c] = (_Float16)val;
    if (interior) yint[((size_t)b * HW_ + (i * Wd_ + j)) * C_ + c] = (_Float16)val;
  }
}

// ---------------- generic WMMA GEMM: D[M,N] = A[M,K] * Bt[N,K]^T + bias[M] --------
// Register-blocked: 4 waves / block, each wave owns a 32x32 macro-tile (2x2 WMMA
// accumulators) -> 4 WMMAs per 8 fragment loads per K-step (2 loads/WMMA).
// A row-major f16; Bt pixel-major f16 (row n = column n of B, contiguous over K).
// MODE 0: f16 out transposed (N,M)          (q_t, k_t)
// MODE 1: f16 out channel-major (M,N)       (v_cm)
// MODE 2: f16 out transposed + ReLU6        (MLP hidden)
// MODE 3: f32 out (M,N) = Res + acc + bias  (final MLP + residual)
template <int MODE>
__global__ void __launch_bounds__(128)
gemm_kernel(const _Float16* __restrict__ A, const _Float16* __restrict__ Bt,
            const float* __restrict__ bias, void* __restrict__ Out,
            const float* __restrict__ Res,
            int M, int N, int K, long sB, long sO, long sR) {
  int wave = threadIdx.x >> 5;
  int wm = wave >> 1, wn = wave & 1;                 // 2x2 waves per block
  int m0 = blockIdx.x * 64 + wm * 32;                // wave macro-tile origin
  int n0 = blockIdx.y * 64 + wn * 32;
  int b = blockIdx.z;
  int lane = threadIdx.x & 31;
  int lr = lane & 15, lhi = lane >> 4;
  // A 16x32 f16 fragment: lane<16 holds K {0..7,16..23}, lane>=16 holds K {8..15,24..31}
  const _Float16* Ar0 = A + (size_t)(m0 + lr) * K + (lhi ? 8 : 0);
  const _Float16* Ar1 = Ar0 + (size_t)16 * K;
  // B 32x16 f16 fragment: lane<16 holds K {0..7,16..23} of column n, lane>=16 holds K {8..15,24..31}
  const _Float16* Br0 = Bt + (size_t)b * sB + (size_t)(n0 + lr) * K + (lhi ? 8 : 0);
  const _Float16* Br1 = Br0 + (size_t)16 * K;
  v8f acc00 = {}, acc01 = {}, acc10 = {}, acc11 = {};
  for (int k = 0; k < K; k += 32) {
    v16h a0 = mk16(*(const f16x8*)(Ar0 + k), *(const f16x8*)(Ar0 + k + 16));
    v16h a1 = mk16(*(const f16x8*)(Ar1 + k), *(const f16x8*)(Ar1 + k + 16));
    v16h b0 = mk16(*(const f16x8*)(Br0 + k), *(const f16x8*)(Br0 + k + 16));
    v16h b1 = mk16(*(const f16x8*)(Br1 + k), *(const f16x8*)(Br1 + k + 16));
    acc00 = wmma16(a0, b0, acc00);
    acc01 = wmma16(a0, b1, acc01);
    acc10 = wmma16(a1, b0, acc10);
    acc11 = wmma16(a1, b1, acc11);
  }
  // D layout per tile: VGPR rI, lane l -> M = rI + (l>=16 ? 8 : 0), N = l&15
  v8f accs[2][2] = {{acc00, acc01}, {acc10, acc11}};
  #pragma unroll
  for (int dm = 0; dm < 2; ++dm) {
    #pragma unroll
    for (int dn = 0; dn < 2; ++dn) {
      #pragma unroll
      for (int rI = 0; rI < 8; ++rI) {
        int m = m0 + dm * 16 + rI + lhi * 8;
        int n = n0 + dn * 16 + lr;
        float v = accs[dm][dn][rI] + bias[m];
        if (MODE == 0) {
          ((_Float16*)Out)[(size_t)b * sO + (size_t)n * M + m] = (_Float16)v;
        } else if (MODE == 1) {
          ((_Float16*)Out)[(size_t)b * sO + (size_t)m * N + n] = (_Float16)v;
        } else if (MODE == 2) {
          v = fminf(fmaxf(v, 0.0f), 6.0f);
          ((_Float16*)Out)[(size_t)b * sO + (size_t)n * M + m] = (_Float16)v;
        } else {
          ((float*)Out)[(size_t)b * sO + (size_t)m * N + n] =
              Res[(size_t)b * sR + (size_t)m * N + n] + v;
        }
      }
    }
  }
}

// ---------------- attention: softmax over 576 distinct keys + log-multiplicity bias --
// qt:(B,256,C) pixel-major, kt:(B,576,C) pixel-major, vcm:(B,C,576) channel-major
// out:(B,C,256) f32
__global__ void __launch_bounds__(128)
attn_kernel(const _Float16* __restrict__ qt, const _Float16* __restrict__ kt,
            const _Float16* __restrict__ vcm, float* __restrict__ out) {
  __shared__ __align__(16) float    S[16][P_];     // energies + log m
  __shared__ __align__(16) _Float16 Pm[16][P_];    // exp(S - rowmax), f16
  __shared__ float scl[16];                        // 1 / rowsum
  int q0 = blockIdx.x * 16;
  int n  = blockIdx.y;
  int b  = blockIdx.z;
  int wave = threadIdx.x >> 5, lane = threadIdx.x & 31;
  int lr = lane & 15, lhi = lane >> 4;

  // ---- pass A: S = Q K^T + log(multiplicity) ----
  const _Float16* qrow = qt + (size_t)(b * HW_ + q0 + lr) * C_ + n * HD_ + (lhi ? 8 : 0);
  for (int t = wave; t < P_ / 16; t += 4) {
    int key = t * 16 + lr;
    const _Float16* krow = kt + (size_t)(b * P_ + key) * C_ + n * HD_ + (lhi ? 8 : 0);
    v8f acc = {};
    #pragma unroll
    for (int kk = 0; kk < HD_; kk += 32) {
      f16x8 alo = *(const f16x8*)(qrow + kk);
      f16x8 ahi = *(const f16x8*)(qrow + kk + 16);
      f16x8 blo = *(const f16x8*)(krow + kk);
      f16x8 bhi = *(const f16x8*)(krow + kk + 16);
      acc = wmma16(mk16(alo, ahi), mk16(blo, bhi), acc);
    }
    int rr = key / HP_, ss = key % HP_;
    int cr = imin(imin(rr, 23 - rr), 7) + 1;   // unfold multiplicity per row
    int cs = imin(imin(ss, 23 - ss), 7) + 1;
    float lbias = __logf((float)(cr * cs));
    #pragma unroll
    for (int rI = 0; rI < 8; ++rI) S[rI + lhi * 8][key] = acc[rI] + lbias;
  }
  __syncthreads();

  // ---- softmax (wave 0: 2 lanes per query row) ----
  if (wave == 0) {
    int row = lane >> 1, half = lane & 1;
    int base = half * (P_ / 2);
    float mx = -3.0e38f;
    for (int jj = 0; jj < P_ / 2; ++jj) mx = fmaxf(mx, S[row][base + jj]);
    mx = fmaxf(mx, __shfl_xor(mx, 1, 32));
    float sm = 0.0f;
    for (int jj = 0; jj < P_ / 2; ++jj) {
      float e = __expf(S[row][base + jj] - mx);
      sm += e;
      Pm[row][base + jj] = (_Float16)e;
    }
    sm += __shfl_xor(sm, 1, 32);
    if (half == 0) scl[row] = 1.0f / sm;
  }
  __syncthreads();

  // ---- pass B: out^T[hd, q] = V[hd, key] * P^T[key, q], scaled by 1/rowsum ----
  for (int ot = wave; ot < HD_ / 16; ot += 4) {
    int hd0 = ot * 16;
    const _Float16* vrow = vcm + (size_t)(b * C_ + n * HD_ + hd0 + lr) * P_ + (lhi ? 8 : 0);
    const _Float16* prow = &Pm[lr][lhi ? 8 : 0];   // B column = query lr, keys in table order
    v8f acc = {};
    #pragma unroll
    for (int kk = 0; kk < P_; kk += 32) {
      f16x8 alo = *(const f16x8*)(vrow + kk);
      f16x8 ahi = *(const f16x8*)(vrow + kk + 16);
      f16x8 blo = *(const f16x8*)(prow + kk);
      f16x8 bhi = *(const f16x8*)(prow + kk + 16);
      acc = wmma16(mk16(alo, ahi), mk16(blo, bhi), acc);
    }
    float sc = scl[lr];
    #pragma unroll
    for (int rI = 0; rI < 8; ++rI) {
      int ch = n * HD_ + hd0 + rI + lhi * 8;
      out[(size_t)(b * C_ + ch) * HW_ + q0 + lr] = acc[rI] * sc;
    }
  }
}

// ---------------- residual1 + BN2 (both f32 carrier and pixel-major f16) -----------
__global__ void resid_bn2_kernel(const float* __restrict__ x, const float* __restrict__ att,
                                 const float* __restrict__ gamma,
                                 const float* __restrict__ g1, const float* __restrict__ b1,
                                 const float* __restrict__ m1, const float* __restrict__ v1,
                                 const float* __restrict__ g2, const float* __restrict__ b2,
                                 const float* __restrict__ m2, const float* __restrict__ v2,
                                 float* __restrict__ x1, _Float16* __restrict__ bn2t) {
  int idx = blockIdx.x * blockDim.x + threadIdx.x;
  if (idx >= B_ * C_ * HW_) return;
  int b  = idx / (C_ * HW_);
  int c  = (idx / HW_) % C_;
  int pq = idx % HW_;
  float s1 = g1[c] * rsqrtf(v1[c] + 1e-5f);
  float sh1 = b1[c] - m1[c] * s1;
  float xv = x[idx];
  // x1 = x + (gamma*attn_out + BN1(x))   [since _swa returns gamma*out + bn_input]
  float v = xv + gamma[0] * att[idx] + (xv * s1 + sh1);
  x1[idx] = v;
  float s2 = g2[c] * rsqrtf(v2[c] + 1e-5f);
  float sh2 = b2[c] - m2[c] * s2;
  bn2t[((size_t)b * HW_ + pq) * C_ + c] = (_Float16)(v * s2 + sh2);
}

extern "C" void kernel_launch(void* const* d_in, const int* in_sizes, int n_in,
                              void* d_out, int out_size, void* d_ws, size_t ws_size,
                              hipStream_t stream) {
  (void)in_sizes; (void)n_in; (void)out_size; (void)ws_size;
  const float* x    = (const float*)d_in[0];
  const float* bn1g = (const float*)d_in[1];
  const float* bn1b = (const float*)d_in[2];
  const float* bn1m = (const float*)d_in[3];
  const float* bn1v = (const float*)d_in[4];
  const float* Wq   = (const float*)d_in[5];
  const float* bq   = (const float*)d_in[6];
  const float* Wk   = (const float*)d_in[7];
  const float* bk   = (const float*)d_in[8];
  const float* Wv   = (const float*)d_in[9];
  const float* bv   = (const float*)d_in[10];
  const float* gam  = (const float*)d_in[11];
  const float* bn2g = (const float*)d_in[12];
  const float* bn2b = (const float*)d_in[13];
  const float* bn2m = (const float*)d_in[14];
  const float* bn2v = (const float*)d_in[15];
  const float* W1   = (const float*)d_in[16];
  const float* b1   = (const float*)d_in[17];
  const float* W2   = (const float*)d_in[18];
  const float* b2   = (const float*)d_in[19];

  char* ws = (char*)d_ws;
  _Float16* Wqh  = (_Float16*)ws; ws += (size_t)C_ * C_ * sizeof(_Float16);
  _Float16* Wkh  = (_Float16*)ws; ws += (size_t)C_ * C_ * sizeof(_Float16);
  _Float16* Wvh  = (_Float16*)ws; ws += (size_t)C_ * C_ * sizeof(_Float16);
  _Float16* W1h  = (_Float16*)ws; ws += (size_t)HID_ * C_ * sizeof(_Float16);
  _Float16* W2h  = (_Float16*)ws; ws += (size_t)C_ * HID_ * sizeof(_Float16);
  _Float16* ypad = (_Float16*)ws; ws += (size_t)B_ * P_ * C_ * sizeof(_Float16);
  _Float16* yint = (_Float16*)ws; ws += (size_t)B_ * HW_ * C_ * sizeof(_Float16);
  _Float16* q_t  = (_Float16*)ws; ws += (size_t)B_ * HW_ * C_ * sizeof(_Float16);
  _Float16* k_t  = (_Float16*)ws; ws += (size_t)B_ * P_ * C_ * sizeof(_Float16);
  _Float16* v_cm = (_Float16*)ws; ws += (size_t)B_ * C_ * P_ * sizeof(_Float16);
  float*    atto = (float*)ws;    ws += (size_t)B_ * C_ * HW_ * sizeof(float);
  float*    x1   = (float*)ws;    ws += (size_t)B_ * C_ * HW_ * sizeof(float);
  _Float16* bn2t = (_Float16*)ws; ws += (size_t)B_ * HW_ * C_ * sizeof(_Float16);
  _Float16* h_t  = (_Float16*)ws; ws += (size_t)B_ * HW_ * HID_ * sizeof(_Float16);

  // weight f32 -> f16
  int nCC = C_ * C_, nHC = HID_ * C_;
  cvt_f16_kernel<<<(nCC + 255) / 256, 256, 0, stream>>>(Wq, Wqh, nCC);
  cvt_f16_kernel<<<(nCC + 255) / 256, 256, 0, stream>>>(Wk, Wkh, nCC);
  cvt_f16_kernel<<<(nCC + 255) / 256, 256, 0, stream>>>(Wv, Wvh, nCC);
  cvt_f16_kernel<<<(nHC + 255) / 256, 256, 0, stream>>>(W1, W1h, nHC);
  cvt_f16_kernel<<<(nHC + 255) / 256, 256, 0, stream>>>(W2, W2h, nHC);

  // BN1 + reflect pad
  bn_pad_kernel<<<dim3(P_, B_), 256, 0, stream>>>(x, bn1g, bn1b, bn1m, bn1v, ypad, yint);

  // q/k/v projections (WMMA, 64x64 block tiles / 32x32 wave tiles)
  gemm_kernel<0><<<dim3(C_ / 64, HW_ / 64, B_), 128, 0, stream>>>(
      Wqh, yint, bq, q_t, nullptr, C_, HW_, C_, (long)HW_ * C_, (long)HW_ * C_, 0);
  gemm_kernel<0><<<dim3(C_ / 64, P_ / 64, B_), 128, 0, stream>>>(
      Wkh, ypad, bk, k_t, nullptr, C_, P_, C_, (long)P_ * C_, (long)P_ * C_, 0);
  gemm_kernel<1><<<dim3(C_ / 64, P_ / 64, B_), 128, 0, stream>>>(
      Wvh, ypad, bv, v_cm, nullptr, C_, P_, C_, (long)P_ * C_, (long)C_ * P_, 0);

  // attention
  attn_kernel<<<dim3(HW_ / 16, NH_, B_), 128, 0, stream>>>(q_t, k_t, v_cm, atto);

  // residual 1 + BN2
  resid_bn2_kernel<<<(B_ * C_ * HW_ + 255) / 256, 256, 0, stream>>>(
      x, atto, gam, bn1g, bn1b, bn1m, bn1v, bn2g, bn2b, bn2m, bn2v, x1, bn2t);

  // MLP: hidden (ReLU6) then output fused with residual 2 into d_out
  gemm_kernel<2><<<dim3(HID_ / 64, HW_ / 64, B_), 128, 0, stream>>>(
      W1h, bn2t, b1, h_t, nullptr, HID_, HW_, C_, (long)HW_ * C_, (long)HW_ * HID_, 0);
  gemm_kernel<3><<<dim3(C_ / 64, HW_ / 64, B_), 128, 0, stream>>>(
      W2h, h_t, b2, d_out, x1, C_, HW_, HID_, (long)HW_ * HID_, (long)C_ * HW_, (long)C_ * HW_);
}
